// GCN_34411277976094
// MI455X (gfx1250) — matrix operand order repaired and checked by reading, not compiled, not run on hardware
//
#include <hip/hip_runtime.h>
#include <hip/hip_bf16.h>

typedef __attribute__((ext_vector_type(16))) _Float16 v16h;
typedef __attribute__((ext_vector_type(8)))  _Float16 v8h;
typedef __attribute__((ext_vector_type(8)))  float    v8f;

#define KCHEB    20
#define NP       1860      // 62*30 real output features
#define NTT      118       // padded N tiles (2-tile pairs per iteration)
#define NPPAD    1888      // 118 * 16
#define NIT      59        // main-loop iterations (2 tiles each)
#define KQ       310       // 62*5 input features
#define KT_TILES 10        // K tiles of 32 (K padded to 320)
#define MT       128       // batch rows per workgroup (8 waves x 16)
#define XSTR     328       // LDS row stride in halves
#define TILE_B   10240     // bytes of one W tile (10*512 halves)
#define PAIR_B   20480     // bytes of one staged pair of W tiles

// ---- workspace layout (bytes) ----
#define OFF_PART 0          // 1024 * 10 floats = 40960
#define OFF_AC   40960      // a[5], c[5]
#define OFF_SUP  41088      // 20*3844 floats = 307520 -> ends 348608
#define OFF_BIAS 348672     // 1888 floats = 7552 -> ends 356224
#define OFF_WT   356352     // 118*10*512 halves = 1208320 B -> ends 1564672
#define OFF_FCP  1564672    // 1888*4 floats = 30208 -> ends 1594880

// ---------------- BN statistics: stage 1 -----------------------------------------
__global__ __launch_bounds__(256) void k_bn_partial(const float* __restrict__ x,
                                                    float* __restrict__ part)
{
    __shared__ float red[10 * 256];
    const int t = threadIdx.x;
    const float* slab = x + (size_t)blockIdx.x * 64 * KQ;
    float s[5], q[5];
#pragma unroll
    for (int f = 0; f < 5; ++f) { s[f] = 0.f; q[f] = 0.f; }
#pragma unroll
    for (int f = 0; f < 5; ++f) {
        for (int it = 0; it < 16; ++it) {
            int g = t + 256 * it;
            if (g < 3968) {
                float v = slab[f + 5 * g];
                s[f] += v;
                q[f] += v * v;
            }
        }
    }
#pragma unroll
    for (int f = 0; f < 5; ++f) { red[f * 256 + t] = s[f]; red[(5 + f) * 256 + t] = q[f]; }
    __syncthreads();
    for (int st = 128; st > 0; st >>= 1) {
        if (t < st) {
#pragma unroll
            for (int j = 0; j < 10; ++j) red[j * 256 + t] += red[j * 256 + t + st];
        }
        __syncthreads();
    }
    if (t == 0) {
#pragma unroll
        for (int j = 0; j < 10; ++j) part[(size_t)blockIdx.x * 10 + j] = red[j * 256];
    }
}

// ---------------- BN statistics: stage 2 -----------------------------------------
__global__ __launch_bounds__(256) void k_bn_final(const float* __restrict__ part, int nblk,
                                                  const float* __restrict__ gamma,
                                                  const float* __restrict__ beta,
                                                  float ntot, float* __restrict__ ac)
{
    __shared__ float red[10 * 256];
    const int t = threadIdx.x;
    float acc[10];
#pragma unroll
    for (int j = 0; j < 10; ++j) acc[j] = 0.f;
    for (int i = t; i < nblk; i += 256)
#pragma unroll
        for (int j = 0; j < 10; ++j) acc[j] += part[(size_t)i * 10 + j];
#pragma unroll
    for (int j = 0; j < 10; ++j) red[j * 256 + t] = acc[j];
    __syncthreads();
    for (int st = 128; st > 0; st >>= 1) {
        if (t < st) {
#pragma unroll
            for (int j = 0; j < 10; ++j) red[j * 256 + t] += red[j * 256 + t + st];
        }
        __syncthreads();
    }
    if (t == 0) {
#pragma unroll
        for (int f = 0; f < 5; ++f) {
            float S = red[f * 256], Q = red[(5 + f) * 256];
            float mean = S / ntot;
            float var  = Q / ntot - mean * mean;
            float a = gamma[f] * rsqrtf(var + 1e-5f);
            ac[f]     = a;
            ac[5 + f] = beta[f] - mean * a;
        }
    }
}

// ---------------- Laplacian + Chebyshev supports (serial recurrence, 1 WG) -------
__global__ __launch_bounds__(1024) void k_cheb(const float* __restrict__ A,
                                               float* __restrict__ sup)
{
    __shared__ float Ls[3844], Pb[3844], Qb[3844];
    __shared__ float dinv[62];
    const int t = threadIdx.x;
    for (int e = t; e < 3844; e += 1024) {
        int n = e / 62, m = e % 62;
        float v = 0.f;
        if (n != m) v = fmaxf(A[e], 0.f) + fmaxf(A[m * 62 + n], 0.f);
        Pb[e] = v;
    }
    __syncthreads();
    if (t < 62) {
        float d = 0.f;
        for (int m = 0; m < 62; ++m) d += Pb[t * 62 + m];
        dinv[t] = rsqrtf(d + 1e-10f);
    }
    __syncthreads();
    for (int e = t; e < 3844; e += 1024) {
        int n = e / 62, m = e % 62;
        float eye = (n == m) ? 1.f : 0.f;
        float L = eye - dinv[n] * Pb[e] * dinv[m];
        Ls[e] = 2.f * L / 2.0f - eye;        // LMAX = 2
    }
    __syncthreads();
    for (int e = t; e < 3844; e += 1024) {
        int n = e / 62, m = e % 62;
        float eye = (n == m) ? 1.f : 0.f;
        sup[e] = eye;
        sup[3844 + e] = Ls[e];
        Qb[e] = eye;
        Pb[e] = Ls[e];
    }
    __syncthreads();
    float* P = Pb;
    float* Q = Qb;
    for (int k = 2; k < KCHEB; ++k) {
        for (int e = t; e < 3844; e += 1024) {
            int n = e / 62, m = e % 62;
            float s = 0.f;
            for (int u = 0; u < 62; ++u) s += Ls[n * 62 + u] * P[u * 62 + m];
            float nv = 2.f * s - Q[e];
            Q[e] = nv;
            sup[(size_t)k * 3844 + e] = nv;
        }
        __syncthreads();
        float* tmp = P; P = Q; Q = tmp;
    }
}

// ---------------- Build W in WMMA-B-fragment-packed f16 layout -------------------
__global__ __launch_bounds__(256) void k_wbuild(const float* __restrict__ sup,
                                                const float* __restrict__ cw,
                                                const float* __restrict__ ac,
                                                _Float16* __restrict__ wt)
{
    const int tile = blockIdx.x;                  // nt * 10 + kt
    const int nt = tile / KT_TILES, kt = tile % KT_TILES;
    for (int e = threadIdx.x; e < 512; e += 256) {
        int l = e >> 4, j = e & 15;
        int p = nt * 16 + (l & 15);
        int K = kt * 32 + ((l & 16) ? 8 : 0) + (j & 7) + ((j & 8) ? 16 : 0);
        float v = 0.f;
        if (p < NP && K < KQ) {
            int n = p / 30, o = p % 30, m = K / 5, i = K % 5;
            float s = 0.f;
#pragma unroll
            for (int k = 0; k < KCHEB; ++k)
                s += sup[(size_t)k * 3844 + n * 62 + m] * cw[k * 150 + i * 30 + o];
            v = s * ac[i];
        }
        wt[(size_t)tile * 512 + e] = (_Float16)v;
    }
}

// ---------------- Folded bias (padded to 1888) + padded fc_w copy ----------------
__global__ __launch_bounds__(256) void k_bias(const float* __restrict__ sup,
                                              const float* __restrict__ cw,
                                              const float* __restrict__ ac,
                                              const float* __restrict__ fcw,
                                              float* __restrict__ bias,
                                              float* __restrict__ fcp)
{
    int p = blockIdx.x * 256 + threadIdx.x;
    if (p >= NPPAD) return;
    float b = 0.f;
    float w0 = 0.f, w1 = 0.f, w2 = 0.f, w3 = 0.f;
    if (p < NP) {
        int n = p / 30, o = p % 30;
        for (int k = 0; k < KCHEB; ++k) {
            float srs = 0.f;
            for (int m = 0; m < 62; ++m) srs += sup[(size_t)k * 3844 + n * 62 + m];
            float cwc = 0.f;
#pragma unroll
            for (int i = 0; i < 5; ++i) cwc += cw[k * 150 + i * 30 + o] * ac[5 + i];
            b += srs * cwc;
        }
        w0 = fcw[p * 4 + 0]; w1 = fcw[p * 4 + 1]; w2 = fcw[p * 4 + 2]; w3 = fcw[p * 4 + 3];
    }
    bias[p] = b;
    fcp[p * 4 + 0] = w0; fcp[p * 4 + 1] = w1; fcp[p * 4 + 2] = w2; fcp[p * 4 + 3] = w3;
}

// ---------------- async global -> LDS pair-of-tiles copy (ASYNCcnt path) ---------
// copies 20480 B: 1280 chunks of 16B = exactly 5 per thread, no divergence
__device__ __forceinline__ void async_pair_load(const _Float16* gsrc, unsigned ldsoff, int t)
{
    unsigned long long a0 = (unsigned long long)(uintptr_t)gsrc;
#pragma unroll
    for (int r = 0; r < 5; ++r) {
        int i = t + r * 256;
        unsigned long long ga = a0 + (unsigned long long)i * 16u;
        unsigned lo = ldsoff + (unsigned)i * 16u;
        asm volatile("global_load_async_to_lds_b128 %0, %1, off"
                     :: "v"(lo), "v"(ga) : "memory");
    }
}

__device__ __forceinline__ void wait_async0()
{
    asm volatile("s_wait_asynccnt 0x0" ::: "memory");
}

// ---------------- Main fused GEMM (WMMA f16) + ReLU + fc -------------------------
__global__ __launch_bounds__(256) void k_main(const float* __restrict__ x,
                                              const _Float16* __restrict__ wt,
                                              const float* __restrict__ bias,
                                              const float* __restrict__ fcp,
                                              const float* __restrict__ fcb,
                                              float* __restrict__ out)
{
    __shared__ __align__(32) unsigned char smem[MT * XSTR * 2];   // 83,968 B
    _Float16* xs  = (_Float16*)smem;                 // phase 1: x tile (f16)
    float*    red = (float*)(smem + 2 * PAIR_B);     // phase 2: fc reduce (32 KB)

    const int t  = threadIdx.x;
    const int b0 = blockIdx.x * MT;

    // ---- stage raw x tile (128 x 310 f32) into LDS as f16 ----
    const float4* slab = (const float4*)(x + (size_t)b0 * KQ);
    for (int idx = t; idx < (MT * KQ) / 4; idx += 256) {
        float4 v = slab[idx];
        int e0 = idx * 4;
        int r  = e0 / KQ;
        int c  = e0 - r * KQ;
        float vv[4] = {v.x, v.y, v.z, v.w};
#pragma unroll
        for (int i = 0; i < 4; ++i) {
            int cc = c + i, rr = r;
            if (cc >= KQ) { cc -= KQ; rr += 1; }
            xs[rr * XSTR + cc] = (_Float16)vv[i];
        }
    }
    for (int idx = t; idx < MT * 18; idx += 256) {    // zero K-pad 310..327
        int r = idx / 18, c = KQ + idx % 18;
        xs[r * XSTR + c] = (_Float16)0.f;
    }
    __syncthreads();

    const int wv = t >> 5;
    const int l  = t & 31;
    const int pl = l & 15;

    // ---- 10 A fragments (16x32 f16, ISA layout) into registers ----
    v16h afrag[KT_TILES];
    {
        const _Float16* row = xs + (size_t)(wv * 16 + (l & 15)) * XSTR;
#pragma unroll
        for (int kt = 0; kt < KT_TILES; ++kt) {
            int base = kt * 32 + ((l & 16) ? 8 : 0);
            v8h lo = *(const v8h*)(row + base);
            v8h hi = *(const v8h*)(row + base + 16);
            afrag[kt] = __builtin_shufflevector(lo, hi, 0, 1, 2, 3, 4, 5, 6, 7,
                                                        8, 9, 10, 11, 12, 13, 14, 15);
        }
    }
    __syncthreads();   // xs dead -> smem reused as double-buffered W tile pairs

    const unsigned wbufOff = (unsigned)(uintptr_t)smem;   // LDS offset of buffer 0

    float acc[8][4];
#pragma unroll
    for (int r = 0; r < 8; ++r)
#pragma unroll
        for (int j = 0; j < 4; ++j) acc[r][j] = 0.f;

    // prologue: async-stage tile pair 0 into buffer 0
    async_pair_load(wt, wbufOff, t);
    wait_async0();
    __syncthreads();

    const float4* fcp4 = (const float4*)fcp;

    for (int it = 0; it < NIT; ++it) {
        if (it + 1 < NIT)                                // prefetch next pair
            async_pair_load(wt + (size_t)(it + 1) * (PAIR_B / 2),
                            wbufOff + (unsigned)((it + 1) & 1) * PAIR_B, t);

        const _Float16* base = (const _Float16*)(smem + (it & 1) * PAIR_B);

#pragma unroll
        for (int hf = 0; hf < 2; ++hf) {
            const int p = (2 * it + hf) * 16 + pl;
            const float bv = bias[p];
            v8f c = {bv, bv, bv, bv, bv, bv, bv, bv};

            const _Float16* wb = base + hf * (TILE_B / 2);

            // depth-5 software pipeline: >=4 ds_load pairs always in flight
            v16h b[5];
#pragma unroll
            for (int k = 0; k < 5; ++k)
                b[k] = *(const v16h*)(wb + (size_t)k * 512 + (size_t)l * 16);
#pragma unroll
            for (int kt = 0; kt < KT_TILES; ++kt) {
                c = __builtin_amdgcn_wmma_f32_16x16x32_f16(false, afrag[kt],
                                                           false, b[kt % 5],
                                                           (short)0, c, false, false);
                if (kt + 5 < KT_TILES)
                    b[kt % 5] = *(const v16h*)(wb + (size_t)(kt + 5) * 512 + (size_t)l * 16);
            }

            float4 fw = fcp4[p];
#pragma unroll
            for (int r = 0; r < 8; ++r) {
                float v = fmaxf(c[r], 0.f);
                acc[r][0] += v * fw.x;
                acc[r][1] += v * fw.y;
                acc[r][2] += v * fw.z;
                acc[r][3] += v * fw.w;
            }
        }

        wait_async0();          // next pair resident (hidden behind 20 WMMAs)
        __syncthreads();        // all waves done reading current buffer
    }

    // ---- cross-lane reduce (16 lanes share each output row) via LDS ----
    {
        int half = l >> 4;
#pragma unroll
        for (int r = 0; r < 8; ++r) {
            int m = half * 8 + r;
            float* dst = red + (size_t)wv * 1024 + m * 64 + pl * 4;
            dst[0] = acc[r][0]; dst[1] = acc[r][1]; dst[2] = acc[r][2]; dst[3] = acc[r][3];
        }
    }
    __syncthreads();
    for (int o = t; o < 512; o += 256) {
        int wave = o >> 6, rem = o & 63, m = rem >> 2, j = rem & 3;
        float s = 0.f;
#pragma unroll
        for (int q = 0; q < 16; ++q) s += red[(size_t)wave * 1024 + m * 64 + q * 4 + j];
        out[(size_t)(b0 + wave * 16 + m) * 4 + j] = s + fcb[j];
    }
}

// ---------------------------------------------------------------------------------
extern "C" void kernel_launch(void* const* d_in, const int* in_sizes, int n_in,
                              void* d_out, int out_size, void* d_ws, size_t ws_size,
                              hipStream_t stream)
{
    const float* x     = (const float*)d_in[0];   // (B, 62, 5)
    const float* A     = (const float*)d_in[1];   // (62, 62)
    const float* gamma = (const float*)d_in[2];   // (5,)
    const float* beta  = (const float*)d_in[3];   // (5,)
    const float* cw    = (const float*)d_in[4];   // (20, 5, 30)
    const float* fcw   = (const float*)d_in[5];   // (1860, 4)
    const float* fcb   = (const float*)d_in[6];   // (4,)
    float* out = (float*)d_out;

    char* ws = (char*)d_ws;
    float*    part = (float*)(ws + OFF_PART);
    float*    ac   = (float*)(ws + OFF_AC);
    float*    sup  = (float*)(ws + OFF_SUP);
    float*    bias = (float*)(ws + OFF_BIAS);
    _Float16* wt   = (_Float16*)(ws + OFF_WT);
    float*    fcp  = (float*)(ws + OFF_FCP);

    const int B = in_sizes[0] / KQ;               // 65536
    const int nblk1 = B / 64;                     // 1024

    k_bn_partial<<<nblk1, 256, 0, stream>>>(x, part);
    k_bn_final<<<1, 256, 0, stream>>>(part, nblk1, gamma, beta,
                                      (float)((double)B * 62.0), ac);
    k_cheb<<<1, 1024, 0, stream>>>(A, sup);
    k_wbuild<<<NTT * KT_TILES, 256, 0, stream>>>(sup, cw, ac, wt);
    k_bias<<<(NPPAD + 255) / 256, 256, 0, stream>>>(sup, cw, ac, fcw, bias, fcp);
    k_main<<<B / MT, 256, 0, stream>>>(x, wt, bias, fcp, fcb, out);
}
